// MemoryGNN_30966714204588
// MI455X (gfx1250) — matrix operand order repaired and checked by reading, not compiled, not run on hardware
//
#include <hip/hip_runtime.h>
#include <stdint.h>

#define HID 64
#define HEADS 4
#define FOUT 256           // HEADS*HID
#define NEG_SLOPE 0.2f

typedef __attribute__((ext_vector_type(16))) __bf16 v16bf;
typedef __attribute__((ext_vector_type(8)))  float  v8f;

union Frag { uint4 u[2]; v16bf v; };

// ---------- helpers ----------
__device__ __forceinline__ unsigned bf16rne(float f) {
  unsigned u = __float_as_uint(f);
  return (u + 0x7fffu + ((u >> 16) & 1u)) >> 16;
}
__device__ __forceinline__ unsigned pk_bf16(float lo, float hi) {
  return bf16rne(lo) | (bf16rne(hi) << 16);
}
// order-preserving float -> uint key (for atomic segment-max)
__device__ __forceinline__ unsigned fkey(float f) {
  int i = __float_as_int(f);
  return (i >= 0) ? ((unsigned)i | 0x80000000u) : ~(unsigned)i;
}
__device__ __forceinline__ float fdec(unsigned k) {
  unsigned bits = (k & 0x80000000u) ? (k & 0x7fffffffu) : ~k;
  return __uint_as_float(bits);
}
__device__ __forceinline__ float leaky(float v) { return v > 0.f ? v : NEG_SLOPE * v; }

// native L2 near-atomic f32 add (GLOBAL_ATOMIC_ADD_F32) instead of CAS loop
__device__ __forceinline__ void atomAddF(float* p, float v) {
  unsafeAtomicAdd(p, v);
}

// ---------- 1. input transform: h = relu(x @ w_in + b_in) -> bf16 [N,64] ----------
__global__ void k_in_xform(const float* __restrict__ x, const float* __restrict__ w,
                           const float* __restrict__ b, unsigned short* __restrict__ hbf, int N) {
  int idx = blockIdx.x * blockDim.x + threadIdx.x;
  if (idx >= N * HID) return;
  int n = idx >> 6, c = idx & 63;
  float acc = b[c];
#pragma unroll
  for (int k = 0; k < 10; ++k) acc += x[n * 10 + k] * w[k * HID + c];
  acc = acc > 0.f ? acc : 0.f;
  hbf[idx] = (unsigned short)bf16rne(acc);
}

// ---------- 2. pack W[F_in,256] f32 -> bf16 WMMA B-fragment layout ----------
// layout: wpk[((kt*16 + nt)*32 + lane)*8 + r] = pack(W[kt*32+half*16+2r][nt*16+(lane&15)], W[..+1][..])
__global__ void k_pack_w(const float* __restrict__ W, unsigned* __restrict__ wpk) {
  int lane = threadIdx.x;                 // 32 threads
  int nt = blockIdx.x & 15;
  int kt = blockIdx.x >> 4;
  int n  = nt * 16 + (lane & 15);
  int kb = kt * 32 + (lane >> 4) * 16;
  unsigned* dst = wpk + ((size_t)blockIdx.x * 32 + lane) * 8;
#pragma unroll
  for (int r = 0; r < 8; ++r) {
    int k = kb + 2 * r;
    dst[r] = pk_bf16(W[(size_t)k * FOUT + n], W[(size_t)(k + 1) * FOUT + n]);
  }
}

// ---------- 3. WMMA GEMM: xl[N,256] = h_bf16[N,F_in] @ W ----------
// one block = 8 waves = one 16-row M tile; wave w owns col-tiles 2w, 2w+1
__global__ void k_gemm(const unsigned short* __restrict__ hbf, const unsigned* __restrict__ wpk,
                       float* __restrict__ xl, int N, int F_in) {
  int lane = threadIdx.x & 31, wave = threadIdx.x >> 5;
  int mtile = blockIdx.x;
  int half = lane >> 4;
  int m = mtile * 16 + (lane & 15);
  int mc = m < N ? m : N - 1;
  const uint4* arow = (const uint4*)(hbf + (size_t)mc * F_in);  // 8 bf16 per uint4
  int nkt = F_in >> 5;
  int nt0 = wave * 2, nt1 = nt0 + 1;
  v8f acc0 = {}, acc1 = {};
  for (int kt = 0; kt < nkt; ++kt) {
    Frag A;                                // A 16x32 bf16: lane=M(0..15), halves split K
    A.u[0] = arow[kt * 4 + 2 * half];      // K = kt*32 + half*16 .. +7
    A.u[1] = arow[kt * 4 + 2 * half + 1];  // K = kt*32 + half*16 + 8 .. +15
    const uint4* b0 = (const uint4*)(wpk + ((size_t)(kt * 16 + nt0) * 32 + lane) * 8);
    const uint4* b1 = (const uint4*)(wpk + ((size_t)(kt * 16 + nt1) * 32 + lane) * 8);
    Frag B0, B1;
    B0.u[0] = b0[0]; B0.u[1] = b0[1];
    B1.u[0] = b1[0]; B1.u[1] = b1[1];
    acc0 = __builtin_amdgcn_wmma_f32_16x16x32_bf16(false, A.v, false, B0.v, (short)0, acc0, false, false);
    acc1 = __builtin_amdgcn_wmma_f32_16x16x32_bf16(false, A.v, false, B1.v, (short)0, acc1, false, false);
  }
#pragma unroll
  for (int r = 0; r < 8; ++r) {            // C/D: lane = N col (mod 16), vgpr r = row (+8*half)
    int row = mtile * 16 + half * 8 + r;
    if (row < N) {
      xl[(size_t)row * FOUT + nt0 * 16 + (lane & 15)] = acc0[r];
      xl[(size_t)row * FOUT + nt1 * 16 + (lane & 15)] = acc1[r];
    }
  }
}

// ---------- 4. attention coefficients + segment-state init ----------
__global__ void k_att(const float* __restrict__ xl, const float* __restrict__ aw_s,
                      const float* __restrict__ aw_d, float* __restrict__ a_s,
                      float* __restrict__ a_d, unsigned* __restrict__ emax_u,
                      float* __restrict__ denom, int N) {
  int idx = blockIdx.x * blockDim.x + threadIdx.x;
  if (idx >= N * HEADS) return;
  int n = idx >> 2, h = idx & 3;
  const float* row = xl + (size_t)n * FOUT + h * HID;
  float s = 0.f, d = 0.f;
#pragma unroll
  for (int c = 0; c < HID; ++c) {
    float v = row[c];
    s += v * aw_s[h * HID + c];
    d += v * aw_d[h * HID + c];
  }
  a_s[idx] = s; a_d[idx] = d;
  emax_u[idx] = 0u;        // encodes "below every real float"
  denom[idx]  = 0.f;
}

// ---------- 5. edge pass A: segment max via monotone-key atomicMax ----------
__global__ void k_edge_max(const int* __restrict__ ei, const float* __restrict__ a_s,
                           const float* __restrict__ a_d, unsigned* __restrict__ emax_u,
                           int E, int N) {
  int e = blockIdx.x * blockDim.x + threadIdx.x;
  if (e >= E + N) return;
  int s, d;
  if (e < E) { s = ei[e]; d = ei[E + e]; } else { s = e - E; d = s; }  // implicit self loops
  float4 sv = ((const float4*)a_s)[s];
  float4 dv = ((const float4*)a_d)[d];
  float v0 = leaky(sv.x + dv.x), v1 = leaky(sv.y + dv.y);
  float v2 = leaky(sv.z + dv.z), v3 = leaky(sv.w + dv.w);
  atomicMax(&emax_u[d * 4 + 0], fkey(v0));
  atomicMax(&emax_u[d * 4 + 1], fkey(v1));
  atomicMax(&emax_u[d * 4 + 2], fkey(v2));
  atomicMax(&emax_u[d * 4 + 3], fkey(v3));
}

// ---------- 6. edge pass B: ex = exp(e - max), denom += ex ----------
__global__ void k_edge_expsum(const int* __restrict__ ei, const float* __restrict__ a_s,
                              const float* __restrict__ a_d, const unsigned* __restrict__ emax_u,
                              float* __restrict__ denom, float* __restrict__ ex,
                              int E, int N) {
  int e = blockIdx.x * blockDim.x + threadIdx.x;
  if (e >= E + N) return;
  int s, d;
  if (e < E) { s = ei[e]; d = ei[E + e]; } else { s = e - E; d = s; }
  float4 sv = ((const float4*)a_s)[s];
  float4 dv = ((const float4*)a_d)[d];
  float v[4] = { leaky(sv.x + dv.x), leaky(sv.y + dv.y), leaky(sv.z + dv.z), leaky(sv.w + dv.w) };
#pragma unroll
  for (int h = 0; h < 4; ++h) {
    float m = fdec(emax_u[d * 4 + h]);
    float x = __expf(v[h] - m);
    ex[(size_t)e * 4 + h] = x;
    atomAddF(&denom[d * 4 + h], x);
  }
}

// ---------- 7. edge pass C: out[dst] += alpha * xl[src]  (one wave per edge) ----------
__global__ void k_edge_scatter(const int* __restrict__ ei, const float* __restrict__ xl,
                               const float* __restrict__ ex, const float* __restrict__ denom,
                               float* __restrict__ acc, int E, int N) {
  int lane = threadIdx.x & 31;
  int e = blockIdx.x * 8 + (threadIdx.x >> 5);
  if (e >= E + N) return;
  int s, d;
  if (e < E) { s = ei[e]; d = ei[E + e]; } else { s = e - E; d = s; }
  int h = lane >> 3;                        // 8 lanes per head, 8 channels per lane
  float alpha = ex[(size_t)e * 4 + h] / denom[d * 4 + h];
  const float4* xs = (const float4*)(xl + (size_t)s * FOUT) + lane * 2;
  float4 v0 = xs[0], v1 = xs[1];
  float* out = acc + (size_t)d * FOUT + lane * 8;
  atomAddF(out + 0, v0.x * alpha); atomAddF(out + 1, v0.y * alpha);
  atomAddF(out + 2, v0.z * alpha); atomAddF(out + 3, v0.w * alpha);
  atomAddF(out + 4, v1.x * alpha); atomAddF(out + 5, v1.y * alpha);
  atomAddF(out + 6, v1.z * alpha); atomAddF(out + 7, v1.w * alpha);
}

// ---------- 8. layers 0/1 epilogue: bias + ELU -> bf16 features ----------
__global__ void k_bias_elu_bf16(const float* __restrict__ acc, const float* __restrict__ b,
                                unsigned short* __restrict__ hbf, int N) {
  int idx = blockIdx.x * blockDim.x + threadIdx.x;
  if (idx >= N * FOUT) return;
  float v = acc[idx] + b[idx & 255];
  v = v > 0.f ? v : (__expf(v) - 1.f);
  hbf[idx] = (unsigned short)bf16rne(v);
}

// ---------- 9. layer 2 epilogue: mean over heads + bias -> h2 f32 [N,64] ----------
__global__ void k_mean_bias(const float* __restrict__ acc, const float* __restrict__ b,
                            float* __restrict__ h2, int N) {
  int idx = blockIdx.x * blockDim.x + threadIdx.x;
  if (idx >= N * HID) return;
  int n = idx >> 6, c = idx & 63;
  const float* r = acc + (size_t)n * FOUT;
  h2[idx] = 0.25f * (r[c] + r[64 + c] + r[128 + c] + r[192 + c]) + b[c];
}

// ---------- 10. node decoder ----------
__global__ void k_node_dec(const float* __restrict__ h2, const float* __restrict__ w1,
                           const float* __restrict__ b1, const float* __restrict__ w2,
                           const float* __restrict__ b2, float* __restrict__ out, int N) {
  int n = blockIdx.x * blockDim.x + threadIdx.x;
  if (n >= N) return;
  const float* r = h2 + (size_t)n * HID;
  float hid[32];
#pragma unroll
  for (int j = 0; j < 32; ++j) hid[j] = b1[j];
  for (int c = 0; c < HID; ++c) {
    float v = r[c];
#pragma unroll
    for (int j = 0; j < 32; ++j) hid[j] += v * w1[c * 32 + j];
  }
  float o0 = b2[0], o1 = b2[1], o2 = b2[2];
#pragma unroll
  for (int j = 0; j < 32; ++j) {
    float v = hid[j] > 0.f ? hid[j] : 0.f;
    o0 += v * w2[j * 3 + 0]; o1 += v * w2[j * 3 + 1]; o2 += v * w2[j * 3 + 2];
  }
  out[(size_t)n * 3 + 0] = o0; out[(size_t)n * 3 + 1] = o1; out[(size_t)n * 3 + 2] = o2;
}

// ---------- 11. graph embedding sum (block-reduced) ----------
__global__ void k_gsum(const float* __restrict__ h2, float* __restrict__ gsum, int N) {
  __shared__ float sd[256];
  int t = threadIdx.x, c = t & 63, q = t >> 6;
  int base = blockIdx.x * 1024;
  float p = 0.f;
  for (int n = base + q; n < base + 1024 && n < N; n += 4) p += h2[(size_t)n * HID + c];
  sd[t] = p;
  __syncthreads();
  if (t < 64) atomAddF(&gsum[c], sd[c] + sd[64 + c] + sd[128 + c] + sd[192 + c]);
}

// ---------- 12. graph decoder ----------
__global__ void k_gdec(const float* __restrict__ gsum, const float* __restrict__ w1,
                       const float* __restrict__ b1, const float* __restrict__ w2,
                       const float* __restrict__ b2, float* __restrict__ out, int N) {
  __shared__ float g[64];
  __shared__ float hid[32];
  int t = threadIdx.x;  // 64 threads
  g[t] = gsum[t] / (float)N;
  __syncthreads();
  if (t < 32) {
    float a = b1[t];
    for (int c = 0; c < 64; ++c) a += g[c] * w1[c * 32 + t];
    hid[t] = (a > 0.f ? a : 0.f) * w2[t];
  }
  __syncthreads();
  if (t == 0) {
    float a = b2[0];
    for (int j = 0; j < 32; ++j) a += hid[j];
    out[0] = 1.f / (1.f + __expf(-a));
  }
}

extern "C" void kernel_launch(void* const* d_in, const int* in_sizes, int n_in,
                              void* d_out, int out_size, void* d_ws, size_t ws_size,
                              hipStream_t stream) {
  const float* x     = (const float*)d_in[0];
  const int*   ei    = (const int*)d_in[1];
  const float* w_in  = (const float*)d_in[2];
  const float* b_in  = (const float*)d_in[3];
  const float* gW[3]  = { (const float*)d_in[4],  (const float*)d_in[8],  (const float*)d_in[12] };
  const float* gAS[3] = { (const float*)d_in[5],  (const float*)d_in[9],  (const float*)d_in[13] };
  const float* gAD[3] = { (const float*)d_in[6],  (const float*)d_in[10], (const float*)d_in[14] };
  const float* gB[3]  = { (const float*)d_in[7],  (const float*)d_in[11], (const float*)d_in[15] };
  const float* nd_w1 = (const float*)d_in[16];
  const float* nd_b1 = (const float*)d_in[17];
  const float* nd_w2 = (const float*)d_in[18];
  const float* nd_b2 = (const float*)d_in[19];
  const float* gd_w1 = (const float*)d_in[20];
  const float* gd_b1 = (const float*)d_in[21];
  const float* gd_w2 = (const float*)d_in[22];
  const float* gd_b2 = (const float*)d_in[23];

  const int N  = in_sizes[0] / 10;   // 100000
  const int E  = in_sizes[1] / 2;    // 1600000
  const int ET = E + N;              // with self-loops

  // ---- workspace carve-out (~315 MB) ----
  char* base = (char*)d_ws;
  size_t off = 0;
  auto carve = [&](size_t bytes) -> void* {
    void* p = base + off;
    off = (off + bytes + 255) & ~(size_t)255;
    return p;
  };
  const size_t XL = (size_t)N * FOUT * sizeof(float);
  float*          xl     = (float*)carve(XL);
  float*          acc    = (float*)carve(XL);
  unsigned short* hbf    = (unsigned short*)carve((size_t)N * FOUT * sizeof(unsigned short));
  float*          ex     = (float*)carve((size_t)ET * HEADS * sizeof(float));
  float*          a_s    = (float*)carve((size_t)N * HEADS * sizeof(float));
  float*          a_d    = (float*)carve((size_t)N * HEADS * sizeof(float));
  float*          denom  = (float*)carve((size_t)N * HEADS * sizeof(float));
  unsigned*       emax_u = (unsigned*)carve((size_t)N * HEADS * sizeof(unsigned));
  float*          h2     = (float*)carve((size_t)N * HID * sizeof(float));
  unsigned*       wpk    = (unsigned*)carve((size_t)8 * 16 * 32 * 8 * sizeof(unsigned));
  float*          gsum   = (float*)carve(64 * sizeof(float));

  // ---- input MLP -> bf16 features (stride 64) ----
  k_in_xform<<<(N * HID + 255) / 256, 256, 0, stream>>>(x, w_in, b_in, hbf, N);

  auto gat = [&](int layer, int F_in) {
    k_pack_w<<<(F_in / 32) * 16, 32, 0, stream>>>(gW[layer], wpk);
    k_gemm<<<(N + 15) / 16, 256, 0, stream>>>(hbf, wpk, xl, N, F_in);
    hipMemsetAsync(acc, 0, XL, stream);
    k_att<<<(N * HEADS + 255) / 256, 256, 0, stream>>>(xl, gAS[layer], gAD[layer], a_s, a_d, emax_u, denom, N);
    k_edge_max<<<(ET + 255) / 256, 256, 0, stream>>>(ei, a_s, a_d, emax_u, E, N);
    k_edge_expsum<<<(ET + 255) / 256, 256, 0, stream>>>(ei, a_s, a_d, emax_u, denom, ex, E, N);
    k_edge_scatter<<<(ET + 7) / 8, 256, 0, stream>>>(ei, xl, ex, denom, acc, E, N);
  };

  gat(0, HID);
  k_bias_elu_bf16<<<(N * FOUT + 255) / 256, 256, 0, stream>>>(acc, gB[0], hbf, N);
  gat(1, FOUT);
  k_bias_elu_bf16<<<(N * FOUT + 255) / 256, 256, 0, stream>>>(acc, gB[1], hbf, N);
  gat(2, FOUT);
  k_mean_bias<<<(N * HID + 255) / 256, 256, 0, stream>>>(acc, gB[2], h2, N);

  k_node_dec<<<(N + 255) / 256, 256, 0, stream>>>(h2, nd_w1, nd_b1, nd_w2, nd_b2, (float*)d_out, N);

  hipMemsetAsync(gsum, 0, 64 * sizeof(float), stream);
  k_gsum<<<(N + 1023) / 1024, 256, 0, stream>>>(h2, gsum, N);
  k_gdec<<<1, 64, 0, stream>>>(gsum, gd_w1, gd_b1, gd_w2, gd_b2, (float*)d_out + (size_t)N * 3, N);
}
